// Indexer_43963285242654
// MI455X (gfx1250) — matrix operand order repaired and checked by reading, not compiled, not run on hardware
//
#include <hip/hip_runtime.h>
#include <hip/hip_bf16.h>

// ---------------- problem constants ----------------
constexpr int S    = 2304;
constexpr int DIM  = 4096;
constexpr int NH   = 64;
constexpr int HD   = 128;
constexpr int RD   = 64;
constexpr int QLR  = 1536;
constexpr int TOPK = 2048;
constexpr int EF   = NH * HD;          // 8192
constexpr float LN_EPS = 1e-5f;
// NH^-0.5 * HD^-0.5
constexpr float WSCALE = 0.01104854346f;

using bf16 = __hip_bfloat16;
typedef __attribute__((ext_vector_type(16))) __bf16 v16bf;
typedef __attribute__((ext_vector_type(8)))  float  v8f;

// ---------------- WMMA fragment helpers ----------------
// A fragment (16x32, bf16, row-major source, row stride ld):
//   lane: m = lane&15, half = lane>>4
//   slots 0-7  : K = half*8      .. half*8+7   (contiguous)
//   slots 8-15 : K = 16+half*8   .. 16+half*8+7 (contiguous)
__device__ inline v16bf load_afrag(const bf16* base, int ld, int lane) {
  int m = lane & 15, half = lane >> 4;
  const bf16* p = base + (size_t)m * ld + half * 8;
  union { v16bf v; uint4 q[2]; } u;
  u.q[0] = *reinterpret_cast<const uint4*>(p);
  u.q[1] = *reinterpret_cast<const uint4*>(p + 16);
  return u.v;
}
// B fragment (32x16, bf16). B[k][n] sourced from a row-major [N][K] buffer
// (i.e. B = W^T where W is stored [n][k]):
//   lane: n = lane&15, half = lane>>4; slots 0-7: K = half*8 .. half*8+7, slots 8-15: K = 16+half*8 .. 16+half*8+7
__device__ inline v16bf load_bfrag(const bf16* base, int ld, int lane) {
  int n = lane & 15, half = lane >> 4;
  const bf16* p = base + (size_t)n * ld + half * 8;
  union { v16bf v; uint4 q[2]; } u;
  u.q[0] = *reinterpret_cast<const uint4*>(p);
  u.q[1] = *reinterpret_cast<const uint4*>(p + 16);
  return u.v;
}
__device__ inline v8f wmma_bf16(v16bf a, v16bf b, v8f c) {
  return __builtin_amdgcn_wmma_f32_16x16x32_bf16(false, a, false, b,
                                                 (short)0, c, false, false);
}
// traditional RoPE on a pair, d = RD = 64
__device__ inline void rope_pair(float& e, float& o, int pos, int i) {
  float freq = __expf(-9.210340372f * (2.0f * i) / 64.0f); // 10000^(-2i/64)
  float ang = (float)pos * freq;
  float sn, cs; __sincosf(ang, &sn, &cs);
  float re = e * cs - o * sn;
  float ro = e * sn + o * cs;
  e = re; o = ro;
}

// ---------------- fp32 -> bf16 convert ----------------
__global__ void cvt_kernel(const float* __restrict__ src, bf16* __restrict__ dst, int n) {
  int i = blockIdx.x * 256 + threadIdx.x;
  int stride = gridDim.x * 256;
  for (; i < n; i += stride) dst[i] = __float2bfloat16(src[i]);
}

// ---------------- k = LN(x @ wk^T) + RoPE ----------------
// one block per 16-row s-tile; 8 waves cover e = 0..127
__global__ void __launch_bounds__(256) k_kernel(const bf16* __restrict__ xb,
                                                const bf16* __restrict__ wkb,
                                                const float* __restrict__ knw,
                                                const float* __restrict__ knb,
                                                bf16* __restrict__ kb) {
  __shared__ float kt[16 * 128];
  __shared__ float mu[16], rs[16];
  int tid = threadIdx.x, lane = tid & 31, wave = tid >> 5;
  int s0 = blockIdx.x * 16;
  int e0 = wave * 16;
  v8f acc = {0,0,0,0,0,0,0,0};
  for (int d0 = 0; d0 < DIM; d0 += 32) {
    v16bf a = load_afrag(xb + (size_t)s0 * DIM + d0, DIM, lane);
    v16bf b = load_bfrag(wkb + (size_t)e0 * DIM + d0, DIM, lane);
    acc = wmma_bf16(a, b, acc);
  }
  int n = lane & 15, half = lane >> 4;
  for (int j = 0; j < 8; ++j) kt[(j + 8 * half) * 128 + e0 + n] = acc[j];
  __syncthreads();
  if (tid < 16) {
    float sum = 0.f;
    for (int e = 0; e < HD; ++e) sum += kt[tid * 128 + e];
    float m = sum * (1.0f / HD);
    float v = 0.f;
    for (int e = 0; e < HD; ++e) { float d = kt[tid * 128 + e] - m; v += d * d; }
    mu[tid] = m;
    rs[tid] = rsqrtf(v * (1.0f / HD) + LN_EPS);
  }
  __syncthreads();
  for (int u = tid; u < 16 * 64; u += 256) {
    int row = u >> 6, i2 = u & 63, e = 2 * i2;
    float v0 = (kt[row * 128 + e]     - mu[row]) * rs[row] * knw[e]     + knb[e];
    float v1 = (kt[row * 128 + e + 1] - mu[row]) * rs[row] * knw[e + 1] + knb[e + 1];
    if (i2 < 32) rope_pair(v0, v1, s0 + row, i2);   // RoPE on first RD dims
    size_t idx = (size_t)(s0 + row) * HD + e;
    kb[idx]     = __float2bfloat16(v0);
    kb[idx + 1] = __float2bfloat16(v1);
  }
}

// ---------------- q = qr @ wq_b^T  (2304 x 8192 x 1536) ----------------
__global__ void __launch_bounds__(256) qproj_kernel(const bf16* __restrict__ qrb,
                                                    const bf16* __restrict__ wqb,
                                                    bf16* __restrict__ qb) {
  int tid = threadIdx.x, lane = tid & 31, wave = tid >> 5;
  int s0 = blockIdx.x * 16;
  int e0 = blockIdx.y * 128 + wave * 16;
  v8f acc = {0,0,0,0,0,0,0,0};
  for (int k0 = 0; k0 < QLR; k0 += 32) {
    v16bf a = load_afrag(qrb + (size_t)s0 * QLR + k0, QLR, lane);
    v16bf b = load_bfrag(wqb + (size_t)e0 * QLR + k0, QLR, lane);
    acc = wmma_bf16(a, b, acc);
  }
  int n = lane & 15, half = lane >> 4;
  for (int j = 0; j < 8; ++j) {
    int m = j + 8 * half;
    qb[(size_t)(s0 + m) * EF + e0 + n] = __float2bfloat16(acc[j]);
  }
}

// ---------------- in-place RoPE on q (first RD dims of each head) ----------------
__global__ void rope_q_kernel(bf16* __restrict__ qb) {
  int u = blockIdx.x * 256 + threadIdx.x;      // one pair per thread
  int i = u & 31;
  int h = (u >> 5) & (NH - 1);
  int s = u >> 11;
  if (s >= S) return;
  size_t idx = (size_t)s * EF + h * HD + 2 * i;
  float e = __bfloat162float(qb[idx]);
  float o = __bfloat162float(qb[idx + 1]);
  rope_pair(e, o, s, i);
  qb[idx]     = __float2bfloat16(e);
  qb[idx + 1] = __float2bfloat16(o);
}

// ---------------- weights = x @ wp^T * scale  (2304 x 64) ----------------
__global__ void __launch_bounds__(128) wproj_kernel(const bf16* __restrict__ xb,
                                                    const bf16* __restrict__ wpb,
                                                    float* __restrict__ w) {
  int tid = threadIdx.x, lane = tid & 31, wave = tid >> 5;
  int s0 = blockIdx.x * 16, h0 = wave * 16;
  v8f acc = {0,0,0,0,0,0,0,0};
  for (int d0 = 0; d0 < DIM; d0 += 32) {
    v16bf a = load_afrag(xb + (size_t)s0 * DIM + d0, DIM, lane);
    v16bf b = load_bfrag(wpb + (size_t)h0 * DIM + d0, DIM, lane);
    acc = wmma_bf16(a, b, acc);
  }
  int n = lane & 15, half = lane >> 4;
  for (int j = 0; j < 8; ++j)
    w[(size_t)(s0 + j + 8 * half) * NH + h0 + n] = acc[j] * WSCALE;
}

// ---------------- fused scores + head-reduction ----------------
// summed[s,t] = sum_h relu(q[s,h,:] . k[t,:]) * w[s,h]; only t-tile <= s-tile.
__global__ void __launch_bounds__(256) scores_kernel(const bf16* __restrict__ qb,
                                                     const bf16* __restrict__ kb,
                                                     const float* __restrict__ w,
                                                     float* __restrict__ summed) {
  __shared__ float wt[16 * 64];
  int tid = threadIdx.x, lane = tid & 31, wave = tid >> 5;
  int s0 = blockIdx.x * 16;
  for (int u = tid; u < 16 * 64; u += 256)
    wt[u] = w[(size_t)(s0 + (u >> 6)) * NH + (u & 63)];
  __syncthreads();
  int t_tile = blockIdx.y * 8 + wave;
  if (t_tile > (int)blockIdx.x) return;      // causal: skip masked tiles
  int t0 = t_tile * 16;
  // k-tile (B operands) held in registers across the head loop
  v16bf b0 = load_bfrag(kb + (size_t)t0 * HD +  0, HD, lane);
  v16bf b1 = load_bfrag(kb + (size_t)t0 * HD + 32, HD, lane);
  v16bf b2 = load_bfrag(kb + (size_t)t0 * HD + 64, HD, lane);
  v16bf b3 = load_bfrag(kb + (size_t)t0 * HD + 96, HD, lane);
  v8f sum = {0,0,0,0,0,0,0,0};
  int n = lane & 15, half = lane >> 4;
  const bf16* qrow = qb + (size_t)s0 * EF;
  for (int h = 0; h < NH; ++h) {
    const bf16* ah = qrow + h * HD;
    v8f acc = {0,0,0,0,0,0,0,0};
    acc = wmma_bf16(load_afrag(ah,      EF, lane), b0, acc);
    acc = wmma_bf16(load_afrag(ah + 32, EF, lane), b1, acc);
    acc = wmma_bf16(load_afrag(ah + 64, EF, lane), b2, acc);
    acc = wmma_bf16(load_afrag(ah + 96, EF, lane), b3, acc);
    if (h + 1 < NH) __builtin_prefetch((const void*)(ah + HD), 0, 0);
    for (int j = 0; j < 8; ++j) {
      float sc = acc[j] > 0.f ? acc[j] : 0.f;                 // relu
      sum[j] += sc * wt[(j + 8 * half) * 64 + h];             // weighted head-reduce
    }
  }
  for (int j = 0; j < 8; ++j)
    summed[(size_t)(s0 + j + 8 * half) * S + t0 + n] = sum[j];
}

// ---------------- per-row deterministic radix top-k ----------------
__device__ inline unsigned mono_key(float f) {
  unsigned b = __float_as_uint(f);
  return (b & 0x80000000u) ? ~b : (b | 0x80000000u);
}
__global__ void __launch_bounds__(256) topk_kernel(const float* __restrict__ summed,
                                                   int* __restrict__ out) {
  int s = blockIdx.x, tid = threadIdx.x;
  int V = s + 1;                       // valid (causal) entries
  int* orow = out + (size_t)s * TOPK;
  if (V <= TOPK) {                     // all valid + lowest masked indices
    for (int j = tid; j < TOPK; j += 256) orow[j] = j;
    return;
  }
  __shared__ unsigned hist[256];
  __shared__ unsigned cnt[256];
  __shared__ unsigned sb1, sb2, sK2, sbase;
  const float* row = summed + (size_t)s * S;
  // pass 1: top byte
  hist[tid] = 0; __syncthreads();
  for (int t = tid; t < V; t += 256) atomicAdd(&hist[mono_key(row[t]) >> 24], 1u);
  __syncthreads();
  if (tid == 0) {
    unsigned cum = 0; int b = 255;
    for (; b > 0; --b) { if (cum + hist[b] >= (unsigned)TOPK) break; cum += hist[b]; }
    sb1 = (unsigned)b; sK2 = (unsigned)TOPK - cum;
  }
  __syncthreads();
  unsigned b1 = sb1, K2 = sK2;
  // pass 2: second byte within bucket b1
  hist[tid] = 0; __syncthreads();
  for (int t = tid; t < V; t += 256) {
    unsigned k = mono_key(row[t]);
    if ((k >> 24) == b1) atomicAdd(&hist[(k >> 16) & 255], 1u);
  }
  __syncthreads();
  if (tid == 0) {
    unsigned cum = 0; int b = 255;
    for (; b > 0; --b) { if (cum + hist[b] >= K2) break; cum += hist[b]; }
    sb2 = (unsigned)b;
  }
  __syncthreads();
  unsigned T = (b1 << 8) | sb2;        // 16-bit selection prefix
  // deterministic compaction: strictly-greater prefix first
  unsigned c = 0;
  for (int t = tid; t < V; t += 256) if ((mono_key(row[t]) >> 16) > T) c++;
  cnt[tid] = c; __syncthreads();
  if (tid == 0) {
    unsigned run = 0;
    for (int i = 0; i < 256; ++i) { unsigned x = cnt[i]; cnt[i] = run; run += x; }
    sbase = run;
  }
  __syncthreads();
  unsigned o = cnt[tid];
  for (int t = tid; t < V; t += 256)
    if ((mono_key(row[t]) >> 16) > T) orow[o++] = t;
  __syncthreads();
  unsigned base = sbase;
  c = 0;
  for (int t = tid; t < V; t += 256) if ((mono_key(row[t]) >> 16) == T) c++;
  cnt[tid] = c; __syncthreads();
  if (tid == 0) {
    unsigned run = base;
    for (int i = 0; i < 256; ++i) { unsigned x = cnt[i]; cnt[i] = run; run += x; }
  }
  __syncthreads();
  o = cnt[tid];
  for (int t = tid; t < V; t += 256)
    if ((mono_key(row[t]) >> 16) == T) { if (o < (unsigned)TOPK) orow[o] = t; o++; }
}

// ---------------- launcher ----------------
extern "C" void kernel_launch(void* const* d_in, const int* in_sizes, int n_in,
                              void* d_out, int out_size, void* d_ws, size_t ws_size,
                              hipStream_t stream) {
  const float* x   = (const float*)d_in[0];
  const float* qr  = (const float*)d_in[1];
  /* d_in[2] = mask (unused; causality handled analytically) */
  const float* wqb_f = (const float*)d_in[3];
  const float* wk  = (const float*)d_in[4];
  const float* knw = (const float*)d_in[5];
  const float* knb = (const float*)d_in[6];
  const float* wp  = (const float*)d_in[7];

  char* p = (char*)d_ws;
  size_t off = 0;
  auto carve = [&](size_t bytes) -> void* {
    void* r = p + off;
    off += (bytes + 255) & ~(size_t)255;
    return r;
  };
  bf16*  xb     = (bf16*) carve((size_t)S * DIM * 2);
  bf16*  qrb    = (bf16*) carve((size_t)S * QLR * 2);
  bf16*  wqb    = (bf16*) carve((size_t)EF * QLR * 2);
  bf16*  wkb    = (bf16*) carve((size_t)HD * DIM * 2);
  bf16*  wpb    = (bf16*) carve((size_t)NH * DIM * 2);
  bf16*  qb     = (bf16*) carve((size_t)S * EF * 2);
  bf16*  kb     = (bf16*) carve((size_t)S * HD * 2);
  float* w      = (float*)carve((size_t)S * NH * 4);
  float* summed = (float*)carve((size_t)S * S * 4);
  (void)ws_size; (void)in_sizes; (void)n_in; (void)out_size;

  cvt_kernel<<<2048, 256, 0, stream>>>(x,     xb,  S * DIM);
  cvt_kernel<<<2048, 256, 0, stream>>>(qr,    qrb, S * QLR);
  cvt_kernel<<<4096, 256, 0, stream>>>(wqb_f, wqb, EF * QLR);
  cvt_kernel<<<512,  256, 0, stream>>>(wk,    wkb, HD * DIM);
  cvt_kernel<<<256,  256, 0, stream>>>(wp,    wpb, NH * DIM);

  k_kernel<<<S / 16, 256, 0, stream>>>(xb, wkb, knw, knb, kb);
  qproj_kernel<<<dim3(S / 16, EF / 128), 256, 0, stream>>>(qrb, wqb, qb);
  rope_q_kernel<<<(S * NH * 32) / 256, 256, 0, stream>>>(qb);
  wproj_kernel<<<S / 16, 128, 0, stream>>>(xb, wpb, w);
  scores_kernel<<<dim3(S / 16, 18), 256, 0, stream>>>(qb, kb, w, summed);
  topk_kernel<<<S, 256, 0, stream>>>(summed, (int*)d_out);
}
